// BayesianOutputLayers_23914377904579
// MI455X (gfx1250) — matrix operand order repaired and checked by reading, not compiled, not run on hardware
//
#include <hip/hip_runtime.h>

// ---------------------------------------------------------------------------
// BayesianOutputLayers forward for MI455X (gfx1250, wave32, WMMA).
//   scores = 20 * rsqrt(1 + pi/8 * (x^2 . softplus(sigma_w)^T)) * (x . cls_w^T) + cls_b
//   deltas = x . bbox_w^T + bbox_b
// bf16 WMMA (fp32 accumulate) GEMMs; tiles staged with CDNA5 async
// global->LDS copies (ASYNCcnt), double-buffered LDS, software-pipelined
// with compile-time buffer selection (branch-free steady state).
// ---------------------------------------------------------------------------

#define N_PROP   16384
#define D_IN     1024
#define C_CLS    1231
#define R_BOX    4920
#define C_PAD    1280        // C padded to multiple of 64 (zero rows)
#define R_PAD    4992        // R padded to multiple of 128 (zero rows)
#define SCALE_F  20.0f
#define PI_OVER_8 0.39269908169872414f

#define BK       64          // K-step per LDS stage
#define NSTAGE   (D_IN / BK) // 16 stages
#define LDT      72          // LDS row stride in bf16 elems (36 dwords -> conflict-free)

typedef __attribute__((ext_vector_type(16))) __bf16 v16bf;
typedef __attribute__((ext_vector_type(8)))  float  v8f;

// CDNA5 async global->LDS copy (GLOBAL_LOAD_ASYNC_TO_LDS_B128, ASYNCcnt).
// INST_OFFSET applies to both the LDS and global addresses (ISA 15.18.3).
#define ASYNC_LOAD_B128(ldsaddr, voff, base, OFF)                          \
  asm volatile("global_load_async_to_lds_b128 %0, %1, %2 offset:" #OFF     \
               :: "v"(ldsaddr), "v"(voff), "s"(base) : "memory")

#define WAIT_ASYNCCNT(N) asm volatile("s_wait_asynccnt " #N ::: "memory")

__device__ __forceinline__ unsigned lds_addr_of(const void* p) {
  // generic pointer to LDS: low 32 bits are the LDS byte offset
  return (unsigned)(unsigned long long)p;
}

__device__ __forceinline__ unsigned int f2bf(float f) {
  unsigned int u = __float_as_uint(f);
  u += 0x7FFFu + ((u >> 16) & 1u);          // round-to-nearest-even
  return (u >> 16) & 0xFFFFu;
}

__device__ __forceinline__ float softplus_f(float v) {
  return (v > 20.0f) ? v : log1pf(__expf(v));
}

// ------------------------------- prep kernels ------------------------------

__global__ void prep_x(const float* __restrict__ x,
                       unsigned short* __restrict__ xb,
                       unsigned short* __restrict__ x2b) {
  int gid = blockIdx.x * blockDim.x + threadIdx.x;
  float4 v = reinterpret_cast<const float4*>(x)[gid];
  uint2 a, b;
  a.x = f2bf(v.x)     | (f2bf(v.y)     << 16);
  a.y = f2bf(v.z)     | (f2bf(v.w)     << 16);
  b.x = f2bf(v.x*v.x) | (f2bf(v.y*v.y) << 16);
  b.y = f2bf(v.z*v.z) | (f2bf(v.w*v.w) << 16);
  reinterpret_cast<uint2*>(xb)[gid]  = a;
  reinterpret_cast<uint2*>(x2b)[gid] = b;
}

__global__ void prep_cls_w(const float* __restrict__ cls_w,
                           const float* __restrict__ sigma_w,
                           unsigned short* __restrict__ wc,
                           unsigned short* __restrict__ wvar) {
  int gid = blockIdx.x * blockDim.x + threadIdx.x;  // float4 slot in padded layout
  int row = gid >> 8;                               // D/4 = 256 float4 per row
  uint2 a = {0u, 0u}, b = {0u, 0u};
  if (row < C_CLS) {
    float4 w = reinterpret_cast<const float4*>(cls_w)[gid];
    float4 s = reinterpret_cast<const float4*>(sigma_w)[gid];
    a.x = f2bf(w.x) | (f2bf(w.y) << 16);
    a.y = f2bf(w.z) | (f2bf(w.w) << 16);
    b.x = f2bf(softplus_f(s.x)) | (f2bf(softplus_f(s.y)) << 16);
    b.y = f2bf(softplus_f(s.z)) | (f2bf(softplus_f(s.w)) << 16);
  }
  reinterpret_cast<uint2*>(wc)[gid]   = a;
  reinterpret_cast<uint2*>(wvar)[gid] = b;
}

__global__ void prep_bbox_w(const float* __restrict__ bbox_w,
                            unsigned short* __restrict__ wb) {
  int gid = blockIdx.x * blockDim.x + threadIdx.x;
  int row = gid >> 8;
  uint2 a = {0u, 0u};
  if (row < R_BOX) {
    float4 w = reinterpret_cast<const float4*>(bbox_w)[gid];
    a.x = f2bf(w.x) | (f2bf(w.y) << 16);
    a.y = f2bf(w.z) | (f2bf(w.w) << 16);
  }
  reinterpret_cast<uint2*>(wb)[gid] = a;
}

// --------------------------- fragment LDS loaders --------------------------
// A 16x32 fragment (ISA 7.12.2): lane m = L%16, half = L/16,
//   elems 0..7  -> K = 8*half + e ; elems 8..15 -> K = 16 + 8*half + (e-8)
__device__ __forceinline__ v16bf lds_fragA(const unsigned short* base, int row, int ks, int half) {
  union { uint4 q[2]; v16bf v; } f;
  const uint4* p = reinterpret_cast<const uint4*>(base + row * LDT + ks + half * 8);
  f.q[0] = p[0];
  f.q[1] = p[2];          // +32 bytes = +16 elements
  return f.v;
}

// B 32x16 fragment: lane n = L%16, half = L/16,
//   elems 0..7  -> K = 8*half + e ; elems 8..15 -> K = 16 + 8*half + (e-8)
__device__ __forceinline__ v16bf lds_fragB(const unsigned short* base, int row, int ks, int half) {
  union { uint4 q[2]; v16bf v; } f;
  const uint4* p = reinterpret_cast<const uint4*>(base + row * LDT + ks + half * 8);
  f.q[0] = p[0];
  f.q[1] = p[2];
  return f.v;
}

// ------------------------------ fused cls GEMM -----------------------------
// Block 128x64, 8 waves in 4x2, wave = 32x32 (2x2 WMMA tiles, m and v fused).
__launch_bounds__(256)
__global__ void cls_gemm(const unsigned short* __restrict__ xb,
                         const unsigned short* __restrict__ x2b,
                         const unsigned short* __restrict__ wc,
                         const unsigned short* __restrict__ wvar,
                         const float* __restrict__ cls_b,
                         float* __restrict__ out) {
  __shared__ alignas(16) unsigned short As [2][128 * LDT];
  __shared__ alignas(16) unsigned short As2[2][128 * LDT];
  __shared__ alignas(16) unsigned short Bsc[2][64 * LDT];
  __shared__ alignas(16) unsigned short Bsv[2][64 * LDT];

  const int tid   = threadIdx.x;
  const int lane  = tid & 31;
  const int wave  = tid >> 5;
  const int wrow  = wave & 3;
  const int wcol  = wave >> 2;
  const int laneM = lane & 15;
  const int half  = lane >> 4;

  const int blockM = blockIdx.y * 128;
  const int blockN = blockIdx.x * 64;

  // staging indices: A 32 elems/thread (4 x b128), B 32 elems/thread (2 x b128 each of wc/wvar)
  const int ar = tid >> 1;                 // 0..127
  const int ac = (tid & 1) * 32;           // 0 or 32
  const int br = tid >> 2;                 // 0..63
  const int bc = (tid & 3) * 16;           // 0,16,32,48

  // per-thread LDS byte addresses (both buffers)
  unsigned lA [2], lA2[2], lBc[2], lBv[2];
#pragma unroll
  for (int b = 0; b < 2; ++b) {
    lA [b] = lds_addr_of(&As [b][ar * LDT + ac]);
    lA2[b] = lds_addr_of(&As2[b][ar * LDT + ac]);
    lBc[b] = lds_addr_of(&Bsc[b][br * LDT + bc]);
    lBv[b] = lds_addr_of(&Bsv[b][br * LDT + bc]);
  }
  // global byte offsets (advance BK*2 = 128 B per stage)
  unsigned voffA = (unsigned)(((blockM + ar) * D_IN + ac) * 2);
  unsigned voffB = (unsigned)(((blockN + br) * D_IN + bc) * 2);

  // 12 async b128 per thread per stage; advances voffA/voffB by one stage
#define CLS_ISSUE(buf)                                          \
  do {                                                          \
    ASYNC_LOAD_B128(lA [buf], voffA, xb,   0);                  \
    ASYNC_LOAD_B128(lA [buf], voffA, xb,  16);                  \
    ASYNC_LOAD_B128(lA [buf], voffA, xb,  32);                  \
    ASYNC_LOAD_B128(lA [buf], voffA, xb,  48);                  \
    ASYNC_LOAD_B128(lA2[buf], voffA, x2b,  0);                  \
    ASYNC_LOAD_B128(lA2[buf], voffA, x2b, 16);                  \
    ASYNC_LOAD_B128(lA2[buf], voffA, x2b, 32);                  \
    ASYNC_LOAD_B128(lA2[buf], voffA, x2b, 48);                  \
    ASYNC_LOAD_B128(lBc[buf], voffB, wc,   0);                  \
    ASYNC_LOAD_B128(lBc[buf], voffB, wc,  16);                  \
    ASYNC_LOAD_B128(lBv[buf], voffB, wvar, 0);                  \
    ASYNC_LOAD_B128(lBv[buf], voffB, wvar,16);                  \
    voffA += 128u; voffB += 128u;                               \
  } while (0)

  v8f acc_m[2][2] = {};
  v8f acc_v[2][2] = {};

  // compute one BK=64 stage from constant buffer bases (folds LDS offsets)
  auto cls_compute = [&](const unsigned short* A_, const unsigned short* A2_,
                         const unsigned short* Bc_, const unsigned short* Bv_) {
#pragma unroll
    for (int ks = 0; ks < BK; ks += 32) {
      v16bf afr[2], a2fr[2], bcfr[2], bvfr[2];
#pragma unroll
      for (int t = 0; t < 2; ++t) {
        const int arow = wrow * 32 + t * 16 + laneM;
        afr[t]  = lds_fragA(A_,  arow, ks, half);
        a2fr[t] = lds_fragA(A2_, arow, ks, half);
        const int brow = wcol * 32 + t * 16 + laneM;
        bcfr[t] = lds_fragB(Bc_, brow, ks, half);
        bvfr[t] = lds_fragB(Bv_, brow, ks, half);
      }
#pragma unroll
      for (int i = 0; i < 2; ++i) {
#pragma unroll
        for (int j = 0; j < 2; ++j) {
          acc_m[i][j] = __builtin_amdgcn_wmma_f32_16x16x32_bf16(
              false, afr[i],  false, bcfr[j], (short)0, acc_m[i][j], false, false);
          acc_v[i][j] = __builtin_amdgcn_wmma_f32_16x16x32_bf16(
              false, a2fr[i], false, bvfr[j], (short)0, acc_v[i][j], false, false);
        }
      }
    }
  };

  // software pipeline over NSTAGE=16 stages, 2 stages / iteration, branch-free body
  CLS_ISSUE(0);                                   // stage 0
#pragma unroll 1
  for (int s = 0; s < NSTAGE - 2; s += 2) {
    CLS_ISSUE(1);                                 // stage s+1
    WAIT_ASYNCCNT(12);
    __syncthreads();
    cls_compute(As[0], As2[0], Bsc[0], Bsv[0]);   // stage s
    __syncthreads();
    CLS_ISSUE(0);                                 // stage s+2
    WAIT_ASYNCCNT(12);
    __syncthreads();
    cls_compute(As[1], As2[1], Bsc[1], Bsv[1]);   // stage s+1
    __syncthreads();
  }
  CLS_ISSUE(1);                                   // stage 15
  WAIT_ASYNCCNT(12);
  __syncthreads();
  cls_compute(As[0], As2[0], Bsc[0], Bsv[0]);     // stage 14
  __syncthreads();
  WAIT_ASYNCCNT(0);
  __syncthreads();
  cls_compute(As[1], As2[1], Bsc[1], Bsv[1]);     // stage 15

  // probit epilogue: 20 * rsqrt(1 + pi/8 * v) * m + bias
#pragma unroll
  for (int j = 0; j < 2; ++j) {
    const int col = blockN + wcol * 32 + j * 16 + laneM;
    if (col < C_CLS) {
      const float bias = cls_b[col];
#pragma unroll
      for (int i = 0; i < 2; ++i) {
        const int row0 = blockM + wrow * 32 + i * 16 + half * 8;
#pragma unroll
        for (int r = 0; r < 8; ++r) {
          float mm = acc_m[i][j][r];
          float vv = acc_v[i][j][r];
          float kk = rsqrtf(1.0f + PI_OVER_8 * vv);
          out[(size_t)(row0 + r) * C_CLS + col] = SCALE_F * kk * mm + bias;
        }
      }
    }
  }
}

// ------------------------------- bbox GEMM ---------------------------------
// Block 128x128, 8 waves in 4x2, wave = 32x64 (2x4 WMMA tiles).
__launch_bounds__(256)
__global__ void bbox_gemm(const unsigned short* __restrict__ xb,
                          const unsigned short* __restrict__ wb,
                          const float* __restrict__ bbox_b,
                          float* __restrict__ out) {
  __shared__ alignas(16) unsigned short As[2][128 * LDT];
  __shared__ alignas(16) unsigned short Bs[2][128 * LDT];

  const int tid   = threadIdx.x;
  const int lane  = tid & 31;
  const int wave  = tid >> 5;
  const int wrow  = wave & 3;      // 4 x 32 rows
  const int wcol  = wave >> 2;     // 2 x 64 cols
  const int laneM = lane & 15;
  const int half  = lane >> 4;

  const int blockM = blockIdx.y * 128;
  const int blockN = blockIdx.x * 128;

  const int ar = tid >> 1;                 // 0..127
  const int ac = (tid & 1) * 32;           // 0 or 32

  unsigned lA[2], lB[2];
#pragma unroll
  for (int b = 0; b < 2; ++b) {
    lA[b] = lds_addr_of(&As[b][ar * LDT + ac]);
    lB[b] = lds_addr_of(&Bs[b][ar * LDT + ac]);
  }
  unsigned voffA = (unsigned)(((blockM + ar) * D_IN + ac) * 2);
  unsigned voffB = (unsigned)(((blockN + ar) * D_IN + ac) * 2);

#define BOX_ISSUE(buf)                                          \
  do {                                                          \
    ASYNC_LOAD_B128(lA[buf], voffA, xb,  0);                    \
    ASYNC_LOAD_B128(lA[buf], voffA, xb, 16);                    \
    ASYNC_LOAD_B128(lA[buf], voffA, xb, 32);                    \
    ASYNC_LOAD_B128(lA[buf], voffA, xb, 48);                    \
    ASYNC_LOAD_B128(lB[buf], voffB, wb,  0);                    \
    ASYNC_LOAD_B128(lB[buf], voffB, wb, 16);                    \
    ASYNC_LOAD_B128(lB[buf], voffB, wb, 32);                    \
    ASYNC_LOAD_B128(lB[buf], voffB, wb, 48);                    \
    voffA += 128u; voffB += 128u;                               \
  } while (0)

  v8f acc[2][4] = {};

  auto box_compute = [&](const unsigned short* A_, const unsigned short* B_) {
#pragma unroll
    for (int ks = 0; ks < BK; ks += 32) {
      v16bf afr[2], bfr[4];
#pragma unroll
      for (int t = 0; t < 2; ++t)
        afr[t] = lds_fragA(A_, wrow * 32 + t * 16 + laneM, ks, half);
#pragma unroll
      for (int t = 0; t < 4; ++t)
        bfr[t] = lds_fragB(B_, wcol * 64 + t * 16 + laneM, ks, half);
#pragma unroll
      for (int i = 0; i < 2; ++i) {
#pragma unroll
        for (int j = 0; j < 4; ++j) {
          acc[i][j] = __builtin_amdgcn_wmma_f32_16x16x32_bf16(
              false, afr[i], false, bfr[j], (short)0, acc[i][j], false, false);
        }
      }
    }
  };

  BOX_ISSUE(0);                                   // stage 0
#pragma unroll 1
  for (int s = 0; s < NSTAGE - 2; s += 2) {
    BOX_ISSUE(1);                                 // stage s+1
    WAIT_ASYNCCNT(8);
    __syncthreads();
    box_compute(As[0], Bs[0]);                    // stage s
    __syncthreads();
    BOX_ISSUE(0);                                 // stage s+2
    WAIT_ASYNCCNT(8);
    __syncthreads();
    box_compute(As[1], Bs[1]);                    // stage s+1
    __syncthreads();
  }
  BOX_ISSUE(1);                                   // stage 15
  WAIT_ASYNCCNT(8);
  __syncthreads();
  box_compute(As[0], Bs[0]);                      // stage 14
  __syncthreads();
  WAIT_ASYNCCNT(0);
  __syncthreads();
  box_compute(As[1], Bs[1]);                      // stage 15

#pragma unroll
  for (int j = 0; j < 4; ++j) {
    const int col = blockN + wcol * 64 + j * 16 + laneM;
    if (col < R_BOX) {
      const float bias = bbox_b[col];
#pragma unroll
      for (int i = 0; i < 2; ++i) {
        const int row0 = blockM + wrow * 32 + i * 16 + half * 8;
#pragma unroll
        for (int r = 0; r < 8; ++r) {
          out[(size_t)(row0 + r) * R_BOX + col] = acc[i][j][r] + bias;
        }
      }
    }
  }
}

// ------------------------------- launcher ----------------------------------
extern "C" void kernel_launch(void* const* d_in, const int* in_sizes, int n_in,
                              void* d_out, int out_size, void* d_ws, size_t ws_size,
                              hipStream_t stream) {
  const float* x       = (const float*)d_in[0];
  const float* cls_w   = (const float*)d_in[1];
  const float* cls_b   = (const float*)d_in[2];
  const float* sigma_w = (const float*)d_in[3];
  const float* bbox_w  = (const float*)d_in[4];
  const float* bbox_b  = (const float*)d_in[5];
  float* out = (float*)d_out;

  // workspace layout (bytes)
  char* ws = (char*)d_ws;
  unsigned short* xb   = (unsigned short*)(ws);                     // N*D bf16      = 33,554,432 B
  unsigned short* x2b  = (unsigned short*)(ws + 33554432);          // N*D bf16      = 33,554,432 B
  unsigned short* wc   = (unsigned short*)(ws + 67108864);          // C_PAD*D bf16  =  2,621,440 B
  unsigned short* wvar = (unsigned short*)(ws + 69730304);          // C_PAD*D bf16  =  2,621,440 B
  unsigned short* wb   = (unsigned short*)(ws + 72351744);          // R_PAD*D bf16  = 10,223,616 B

  prep_x     <<<(N_PROP * D_IN) / 1024, 256, 0, stream>>>(x, xb, x2b);
  prep_cls_w <<<(C_PAD  * D_IN) / 1024, 256, 0, stream>>>(cls_w, sigma_w, wc, wvar);
  prep_bbox_w<<<(R_PAD  * D_IN) / 1024, 256, 0, stream>>>(bbox_w, wb);

  dim3 gc(C_PAD / 64, N_PROP / 128);
  cls_gemm<<<gc, 256, 0, stream>>>(xb, x2b, wc, wvar, cls_b, out);

  dim3 gb(R_PAD / 128, N_PROP / 128);
  bbox_gemm<<<gb, 256, 0, stream>>>(xb, wb, bbox_b, out + (size_t)N_PROP * C_CLS);
}
